// MultiHeadSelfAttention_11381663334438
// MI455X (gfx1250) — compile-verified, matrix operand order checked by reading, not run on hardware
//
#include <hip/hip_runtime.h>
#include <hip/hip_bf16.h>
#include <stdint.h>

typedef __attribute__((ext_vector_type(16))) __bf16 v16bf;
typedef __attribute__((ext_vector_type(8)))  float  v8f;
typedef __attribute__((vector_size(16)))     int    v4i;

#define S_LEN    2048
#define DM       1024
#define NHEADS   16
#define DK       64
#define BATCH    4
#define M_TOT    (BATCH * S_LEN)   // 8192

// ---- CDNA5 async global->LDS path (probe-guarded; builtins confirmed) ----
#if defined(__has_builtin)
#if __has_builtin(__builtin_amdgcn_global_load_async_to_lds_b128) && \
    __has_builtin(__builtin_amdgcn_s_wait_asynccnt)
#define USE_ASYNC_LDS 1
#endif
#endif
#ifndef USE_ASYNC_LDS
#define USE_ASYNC_LDS 0
#endif

#if USE_ASYNC_LDS
// Builtin takes generic v4i* pointers (per clang diagnostic), plus immediate
// byte offset (applied to both addresses) and immediate cpol.
#define ASYNC_CP16(GP, LP, OFF)                                              \
  __builtin_amdgcn_global_load_async_to_lds_b128((v4i*)(GP), (v4i*)(LP),     \
                                                 (OFF), 0)
#define ASYNC_WAIT() __builtin_amdgcn_s_wait_asynccnt(0)
#endif

__device__ __forceinline__ v8f wmma_bf16(v16bf a, v16bf b, v8f c) {
  // D = A(16x32) * B(32x16) + C, fp32 accumulate
  return __builtin_amdgcn_wmma_f32_16x16x32_bf16(false, a, false, b, (short)0, c,
                                                 false, false);
}

// ---------------- fp32 -> bf16 cast ----------------
__global__ void cast_f32_bf16(const float* __restrict__ in,
                              __bf16* __restrict__ out, int n) {
  int i = blockIdx.x * blockDim.x + threadIdx.x;
  int stride = gridDim.x * blockDim.x;
  for (; i < n; i += stride) out[i] = (__bf16)in[i];
}

// ---------------- WMMA GEMM: Y[m][n] = scale * sum_k A[m][k] * W[n][k] ----
// Double-buffered LDS pipeline: next tile's loads are in flight while the
// current tile's 8 WMMAs execute; one barrier per K-step.
// mode 0: bf16 output in [B, H, S, DK] layout (n -> h*64+d)   (Q/K/V proj)
// mode 1: fp32 output [M][N]                                   (o-proj)
__global__ __launch_bounds__(256) void gemm_bf16_nt(
    const __bf16* __restrict__ A, const __bf16* __restrict__ W,
    __bf16* __restrict__ out_bf, float* __restrict__ out_f,
    int M, int N, int Kd, int mode, float scale)
{
  __shared__ __bf16 As[2][128][40];   // 40 = 32 + 8 pad, keeps 16B alignment
  __shared__ __bf16 Bs[2][128][40];

  const int tid  = threadIdx.x;
  const int wave = tid >> 5, lane = tid & 31;
  const int lrow = lane & 15, lhi = lane >> 4;
  const int wm = wave >> 1, wn = wave & 1;       // 4x2 wave grid
  const int m0 = blockIdx.y * 128, n0 = blockIdx.x * 128;
  const int lr = tid >> 1, lh = tid & 1;         // tile loader mapping

  const __bf16* ga = A + (size_t)(m0 + lr) * Kd + lh * 16;
  const __bf16* gb = W + (size_t)(n0 + lr) * Kd + lh * 16;

  const v8f vzero = {};
  v8f acc[2][4];
#pragma unroll
  for (int i = 0; i < 2; ++i)
#pragma unroll
    for (int j = 0; j < 4; ++j) acc[i][j] = vzero;

  const int nk = Kd / 32;

#if USE_ASYNC_LDS
  // prologue: tile 0 via async DMA to LDS
  ASYNC_CP16(ga, &As[0][lr][lh * 16], 0);
  ASYNC_CP16(ga, &As[0][lr][lh * 16], 16);
  ASYNC_CP16(gb, &Bs[0][lr][lh * 16], 0);
  ASYNC_CP16(gb, &Bs[0][lr][lh * 16], 16);
  ASYNC_WAIT();
  __syncthreads();
  for (int kt = 0; kt < nk; ++kt) {
    const int cur = kt & 1, nxt = cur ^ 1;
    if (kt + 1 < nk) {   // issue next tile while computing this one
      const __bf16* gan = ga + (kt + 1) * 32;
      const __bf16* gbn = gb + (kt + 1) * 32;
      ASYNC_CP16(gan, &As[nxt][lr][lh * 16], 0);
      ASYNC_CP16(gan, &As[nxt][lr][lh * 16], 16);
      ASYNC_CP16(gbn, &Bs[nxt][lr][lh * 16], 0);
      ASYNC_CP16(gbn, &Bs[nxt][lr][lh * 16], 16);
    }
    v16bf af[2], bf[4];
#pragma unroll
    for (int i = 0; i < 2; ++i)
      af[i] = *(const v16bf*)&As[cur][wm * 32 + i * 16 + lrow][lhi * 16];
#pragma unroll
    for (int j = 0; j < 4; ++j)
      bf[j] = *(const v16bf*)&Bs[cur][wn * 64 + j * 16 + lrow][lhi * 16];
#pragma unroll
    for (int i = 0; i < 2; ++i)
#pragma unroll
      for (int j = 0; j < 4; ++j)
        acc[i][j] = wmma_bf16(af[i], bf[j], acc[i][j]);
    ASYNC_WAIT();
    __syncthreads();
  }
#else
  // register-pipelined fallback: distinct register quads, loads issued before
  // compute, LDS stores after compute.
  uint4 ra0 = ((const uint4*)ga)[0], ra1 = ((const uint4*)ga)[1];
  uint4 rb0 = ((const uint4*)gb)[0], rb1 = ((const uint4*)gb)[1];
  {
    uint4* da = (uint4*)&As[0][lr][lh * 16]; da[0] = ra0; da[1] = ra1;
    uint4* db = (uint4*)&Bs[0][lr][lh * 16]; db[0] = rb0; db[1] = rb1;
  }
  __syncthreads();
  for (int kt = 0; kt < nk; ++kt) {
    const int cur = kt & 1, nxt = cur ^ 1;
    if (kt + 1 < nk) {
      const uint4* sa = (const uint4*)(ga + (kt + 1) * 32);
      const uint4* sb = (const uint4*)(gb + (kt + 1) * 32);
      ra0 = sa[0]; ra1 = sa[1];
      rb0 = sb[0]; rb1 = sb[1];
    }
    v16bf af[2], bf[4];
#pragma unroll
    for (int i = 0; i < 2; ++i)
      af[i] = *(const v16bf*)&As[cur][wm * 32 + i * 16 + lrow][lhi * 16];
#pragma unroll
    for (int j = 0; j < 4; ++j)
      bf[j] = *(const v16bf*)&Bs[cur][wn * 64 + j * 16 + lrow][lhi * 16];
#pragma unroll
    for (int i = 0; i < 2; ++i)
#pragma unroll
      for (int j = 0; j < 4; ++j)
        acc[i][j] = wmma_bf16(af[i], bf[j], acc[i][j]);
    if (kt + 1 < nk) {
      uint4* da = (uint4*)&As[nxt][lr][lh * 16]; da[0] = ra0; da[1] = ra1;
      uint4* db = (uint4*)&Bs[nxt][lr][lh * 16]; db[0] = rb0; db[1] = rb1;
    }
    __syncthreads();
  }
#endif

  // epilogue: C layout = (row r + 8*lhi, col lrow) per accumulator VGPR r
#pragma unroll
  for (int i = 0; i < 2; ++i)
#pragma unroll
    for (int j = 0; j < 4; ++j)
#pragma unroll
      for (int r = 0; r < 8; ++r) {
        int m = m0 + wm * 32 + i * 16 + r + 8 * lhi;
        int n = n0 + wn * 64 + j * 16 + lrow;
        float v = acc[i][j][r] * scale;
        if (mode == 0) {
          int b = m / S_LEN, s = m % S_LEN;
          int h = n >> 6, d = n & 63;
          out_bf[(((size_t)b * NHEADS + h) * S_LEN + s) * DK + d] = (__bf16)v;
        } else {
          out_f[(size_t)m * N + n] = v;
        }
      }
}

// ---------------- Flash attention: per (b,h,64-row q tile) ----------------
// Q is pre-scaled by 1/sqrt(dk). 4 waves, each wave owns 16 q rows.
__global__ __launch_bounds__(128) void flash_attn(
    const __bf16* __restrict__ Q, const __bf16* __restrict__ K,
    const __bf16* __restrict__ V, __bf16* __restrict__ O)
{
  __shared__ __bf16 Vt[64][40];        // V^T chunk: [d][key], padded rows
  __shared__ __bf16 Pst[4][16][40];    // per-wave P bounce (C->A layout)

  const int tid  = threadIdx.x;
  const int wave = tid >> 5, lane = tid & 31;
  const int lrow = lane & 15, lhi = lane >> 4;
  const int qt = blockIdx.x & 31;        // S/64 = 32 q tiles
  const int bh = blockIdx.x >> 5;        // b*NHEADS + h
  const int h = bh & (NHEADS - 1), b = bh / NHEADS;
  const size_t base = (size_t)bh * S_LEN * DK;
  const __bf16* Qp = Q + base;
  const __bf16* Kp = K + base;
  const __bf16* Vp = V + base;
  const int q0 = qt * 64 + wave * 16;

  // Q A-fragments: lane lrow = row, lhi selects k 0-15 vs 16-31 of each chunk
  v16bf qf[2];
#pragma unroll
  for (int j = 0; j < 2; ++j)
    qf[j] = *(const v16bf*)(Qp + (size_t)(q0 + lrow) * DK + j * 32 + lhi * 16);

  const v8f vzero = {};
  v8f o[4];
#pragma unroll
  for (int n = 0; n < 4; ++n) o[n] = vzero;
  float mrow[8], lsum[8];
#pragma unroll
  for (int r = 0; r < 8; ++r) { mrow[r] = -1e30f; lsum[r] = 0.0f; }

  const int vkey = tid >> 2, vseg = tid & 3;   // V transposer mapping

  for (int kt = 0; kt < S_LEN / 32; ++kt) {
    const int k0 = kt * 32;
    __syncthreads();
    {  // transpose 32 keys x 64 d of V into LDS
      v16bf vv = *(const v16bf*)(Vp + (size_t)(k0 + vkey) * DK + vseg * 16);
#pragma unroll
      for (int e = 0; e < 16; ++e) Vt[vseg * 16 + e][vkey] = vv[e];
    }
    __syncthreads();

    // scores S = Q * K^T for 16q x 32keys (two 16-key column groups)
    v8f s[2]; s[0] = vzero; s[1] = vzero;
#pragma unroll
    for (int g = 0; g < 2; ++g)
#pragma unroll
      for (int j = 0; j < 2; ++j) {
        v16bf kf = *(const v16bf*)(Kp + (size_t)(k0 + g * 16 + lrow) * DK +
                                   j * 32 + lhi * 16);
        s[g] = wmma_bf16(qf[j], kf, s[g]);
      }

    // online softmax (row stats replicated across the 16-lane group)
    float p0[8], p1[8];
#pragma unroll
    for (int r = 0; r < 8; ++r) {
      float v = fmaxf(s[0][r], s[1][r]);
      v = fmaxf(v, __shfl_xor(v, 1));
      v = fmaxf(v, __shfl_xor(v, 2));
      v = fmaxf(v, __shfl_xor(v, 4));
      v = fmaxf(v, __shfl_xor(v, 8));
      float mnew = fmaxf(mrow[r], v);
      float corr = __expf(mrow[r] - mnew);
      mrow[r] = mnew;
      p0[r] = __expf(s[0][r] - mnew);
      p1[r] = __expf(s[1][r] - mnew);
      float ps = p0[r] + p1[r];
      ps += __shfl_xor(ps, 1);
      ps += __shfl_xor(ps, 2);
      ps += __shfl_xor(ps, 4);
      ps += __shfl_xor(ps, 8);
      lsum[r] = lsum[r] * corr + ps;
#pragma unroll
      for (int n = 0; n < 4; ++n) o[n][r] *= corr;
    }

    // bounce P through per-wave LDS: C layout -> A fragment layout
#pragma unroll
    for (int r = 0; r < 8; ++r) {
      Pst[wave][r + 8 * lhi][lrow]      = (__bf16)p0[r];
      Pst[wave][r + 8 * lhi][16 + lrow] = (__bf16)p1[r];
    }
    v16bf pf = *(const v16bf*)&Pst[wave][lrow][lhi * 16];

    // O += P(16x32) * V(32x64): 4 d-column groups
#pragma unroll
    for (int n = 0; n < 4; ++n) {
      v16bf vf = *(const v16bf*)&Vt[n * 16 + lrow][lhi * 16];
      o[n] = wmma_bf16(pf, vf, o[n]);
    }
  }

  // normalize and store concat-head layout [B, S, H*DK] as bf16
#pragma unroll
  for (int r = 0; r < 8; ++r) {
    float inv = 1.0f / lsum[r];
    int q = q0 + r + 8 * lhi;
#pragma unroll
    for (int n = 0; n < 4; ++n) {
      size_t idx = ((size_t)b * S_LEN + q) * DM + h * DK + n * 16 + lrow;
      O[idx] = (__bf16)(o[n][r] * inv);
    }
  }
}

// ---------------- bias + residual + LayerNorm ----------------
__global__ __launch_bounds__(256) void resid_ln(
    const float* __restrict__ x, const float* __restrict__ proj,
    const float* __restrict__ b_o, const float* __restrict__ g,
    const float* __restrict__ be, float* __restrict__ y)
{
  __shared__ float red[256];
  const int row = blockIdx.x;
  const int tid = threadIdx.x;
  float t[4];
  float s = 0.f, s2 = 0.f;
#pragma unroll
  for (int i = 0; i < 4; ++i) {
    int c = tid + i * 256;
    float v = x[(size_t)row * DM + c] + proj[(size_t)row * DM + c] + b_o[c];
    t[i] = v; s += v; s2 += v * v;
  }
  red[tid] = s; __syncthreads();
  for (int off = 128; off > 0; off >>= 1) {
    if (tid < off) red[tid] += red[tid + off];
    __syncthreads();
  }
  float mean = red[0] * (1.0f / DM);
  __syncthreads();
  red[tid] = s2; __syncthreads();
  for (int off = 128; off > 0; off >>= 1) {
    if (tid < off) red[tid] += red[tid + off];
    __syncthreads();
  }
  float var = red[0] * (1.0f / DM) - mean * mean;
  float rstd = rsqrtf(var + 1e-5f);
#pragma unroll
  for (int i = 0; i < 4; ++i) {
    int c = tid + i * 256;
    y[(size_t)row * DM + c] = (t[i] - mean) * rstd * g[c] + be[c];
  }
}

// ---------------- driver ----------------
extern "C" void kernel_launch(void* const* d_in, const int* in_sizes, int n_in,
                              void* d_out, int out_size, void* d_ws, size_t ws_size,
                              hipStream_t stream) {
  const float* x    = (const float*)d_in[0];
  const float* w_q  = (const float*)d_in[1];
  const float* w_k  = (const float*)d_in[2];
  const float* w_v  = (const float*)d_in[3];
  const float* w_o  = (const float*)d_in[4];
  const float* b_o  = (const float*)d_in[5];
  const float* ln_g = (const float*)d_in[6];
  const float* ln_b = (const float*)d_in[7];
  float* out = (float*)d_out;

  char* ws = (char*)d_ws;
  const size_t MB = 1024 * 1024;
  __bf16* xb   = (__bf16*)(ws);             // 16 MB: x bf16 [8192,1024]
  __bf16* wqb  = (__bf16*)(ws + 16 * MB);   // 2 MB each
  __bf16* wkb  = (__bf16*)(ws + 18 * MB);
  __bf16* wvb  = (__bf16*)(ws + 20 * MB);
  __bf16* wob  = (__bf16*)(ws + 22 * MB);
  __bf16* Qb   = (__bf16*)(ws + 24 * MB);   // 16 MB [B,H,S,dk] (pre-scaled)
  __bf16* Kb   = (__bf16*)(ws + 40 * MB);   // 16 MB
  __bf16* Vb   = (__bf16*)(ws + 56 * MB);   // 16 MB
  __bf16* Ob   = (__bf16*)(ws + 72 * MB);   // 16 MB concat-head attn out
  float*  proj = (float*)(ws + 88 * MB);    // 32 MB o-proj output

  cast_f32_bf16<<<2048, 256, 0, stream>>>(x,   xb,  M_TOT * DM);
  cast_f32_bf16<<<1024, 256, 0, stream>>>(w_q, wqb, DM * DM);
  cast_f32_bf16<<<1024, 256, 0, stream>>>(w_k, wkb, DM * DM);
  cast_f32_bf16<<<1024, 256, 0, stream>>>(w_v, wvb, DM * DM);
  cast_f32_bf16<<<1024, 256, 0, stream>>>(w_o, wob, DM * DM);

  dim3 gg(DM / 128, M_TOT / 128);   // 8 x 64 blocks
  gemm_bf16_nt<<<gg, 256, 0, stream>>>(xb, wqb, Qb, nullptr, M_TOT, DM, DM, 0, 0.125f);
  gemm_bf16_nt<<<gg, 256, 0, stream>>>(xb, wkb, Kb, nullptr, M_TOT, DM, DM, 0, 1.0f);
  gemm_bf16_nt<<<gg, 256, 0, stream>>>(xb, wvb, Vb, nullptr, M_TOT, DM, DM, 0, 1.0f);

  flash_attn<<<BATCH * NHEADS * (S_LEN / 64), 128, 0, stream>>>(Qb, Kb, Vb, Ob);

  gemm_bf16_nt<<<gg, 256, 0, stream>>>(Ob, wob, nullptr, proj, M_TOT, DM, DM, 1, 1.0f);

  resid_ln<<<M_TOT, 256, 0, stream>>>(x, proj, b_o, ln_g, ln_b, out);
}
